// att1_84104049590808
// MI455X (gfx1250) — matrix operand order repaired and checked by reading, not compiled, not run on hardware
//
#include <hip/hip_runtime.h>

#define B_ 8
#define T_ 2048
#define H_ 512
#define L_ 8000
#define M_ 1000
#define NEGV (-1e6f)
#define SBT_S 40   // padded stride (elements) for transposed staging; 80B, 16B-aligned, bank-spread

typedef __attribute__((ext_vector_type(16))) __bf16 v16bf;
typedef __attribute__((ext_vector_type(8)))  float  v8f;

union Frag16 {
  v16bf v;
  int4 q[2];
  unsigned short u[16];
};

__device__ __forceinline__ unsigned short f2bf(float f) {
  unsigned int x = __float_as_uint(f);
  x += 0x7FFFu + ((x >> 16) & 1u);      // round-to-nearest-even truncation
  return (unsigned short)(x >> 16);
}

// K offset of element e inside a 16x32 bf16 A-fragment (ISA 7.12.2):
// lanes 0-15 (kh=0): VGPR0..3 -> K 0..7, VGPR4..7 -> K 16..23
// lanes 16-31 (kh=1): +8
__device__ __forceinline__ int a_koff(int e, int kh) {
  int p = e >> 1, j = e & 1;
  int base = (p < 4) ? (p << 1) : (16 + ((p - 4) << 1));
  return base + (kh << 3) + j;
}

__device__ __forceinline__ float leakyf(float x) {
  return x >= 0.0f ? x : 0.01f * x;
}

// ---------------- prep kernels ----------------

__global__ void k_f32_to_bf16(const float* __restrict__ src,
                              unsigned short* __restrict__ dst, int n) {
  int i = blockIdx.x * blockDim.x + threadIdx.x;
  if (i < n) dst[i] = f2bf(src[i]);
}

// gate = bf16(tanh(two_wp_W * p_p_h)), wop_bf = bf16(weighted_outputs_par)
__global__ void k_gate_wop(const float* __restrict__ p_p_h,
                           const float* __restrict__ two_wp_W,
                           const float* __restrict__ wop,
                           unsigned short* __restrict__ gate_bf,
                           unsigned short* __restrict__ wop_bf, int n) {
  int i = blockIdx.x * blockDim.x + threadIdx.x;
  if (i >= n) return;
  int mh = i % (M_ * H_);
  gate_bf[i] = f2bf(tanhf(two_wp_W[mh] * p_p_h[i]));
  wop_bf[i]  = f2bf(wop[i]);
}

// w[l,:] = leaky_relu(des[l,:] @ W^T + b)   (one block per l)
__global__ void k_wlin(const float* __restrict__ des,
                       const float* __restrict__ W,
                       const float* __restrict__ bias,
                       float* __restrict__ w_out) {
  int l = blockIdx.x;
  const float* dr = des + (size_t)l * H_;
  for (int ho = threadIdx.x; ho < H_; ho += blockDim.x) {
    const float* wr = W + (size_t)ho * H_;
    float s = bias[ho];
    for (int h = 0; h < H_; ++h) s = fmaf(dr[h], wr[h], s);
    w_out[(size_t)l * H_ + ho] = leakyf(s);
  }
}

// bvec[l] = leaky_relu(des[l,:] @ b_lin_W + b_lin_b)
__global__ void k_bvec(const float* __restrict__ des,
                       const float* __restrict__ bW,
                       const float* __restrict__ bB,
                       float* __restrict__ bvec) {
  int l = blockIdx.x * blockDim.x + threadIdx.x;
  if (l >= L_) return;
  const float* dr = des + (size_t)l * H_;
  float s = bB[0];
  for (int h = 0; h < H_; ++h) s = fmaf(dr[h], bW[h], s);
  bvec[l] = leakyf(s);
}

// ---------------- fused attention kernel ----------------
// One workgroup = (batch b, 16 consecutive rows of L). 8 wave32s, one WG per WGP.
__launch_bounds__(256, 1)
__global__ void fused_att_kernel(const unsigned short* __restrict__ text_bf,
                                 const int* __restrict__ mask,
                                 const unsigned short* __restrict__ des_bf,
                                 const unsigned short* __restrict__ gate_bf,
                                 const unsigned short* __restrict__ wop_bf,
                                 const float* __restrict__ w_f32,
                                 const float* __restrict__ bvec,
                                 float* __restrict__ out) {
  __shared__ float s_score[16][T_];           // 128 KB; rows reused as s_slp[16][1024]
  __shared__ unsigned short s_desm[16][H_];   // 16 KB: des A-tile, later bf16(m)
  __shared__ unsigned short sBT[H_ * SBT_S];  // 40 KB: transposed 32 x H k-chunk staging
  __shared__ float s_logit[16];

  const int bid  = blockIdx.x;
  const int b    = bid / (L_ / 16);
  const int l0   = (bid % (L_ / 16)) * 16;
  const int tid  = threadIdx.x;
  const int wave = tid >> 5;
  const int lane = tid & 31;
  const int n_l  = lane & 15;   // N (or M-row for A frags) within 16-wide tile
  const int kh   = lane >> 4;   // K-half select

  // ---- load des A-tile, zero logits ----
  for (int i = tid; i < 16 * H_; i += 256) {
    int r = i >> 9, h = i & (H_ - 1);
    s_desm[r][h] = des_bf[(size_t)(l0 + r) * H_ + h];
  }
  if (tid < 16) s_logit[tid] = 0.0f;
  __syncthreads();

  // ---- phase 1: score[16][T] = des_tile @ text[b]^T  (K = H) ----
  for (int ti = 0; ti < 16; ++ti) {
    const int t0 = (wave * 16 + ti) * 16;
    v8f acc = {0.f, 0.f, 0.f, 0.f, 0.f, 0.f, 0.f, 0.f};
    const unsigned short* brow = text_bf + (size_t)(b * T_ + t0 + n_l) * H_;
    if (ti + 1 < 16) {  // pull next tile's row toward the WGP while WMMAs run
      __builtin_prefetch(text_bf + (size_t)(b * T_ + t0 + 16 + n_l) * H_, 0, 1);
    }
    for (int k0 = 0; k0 < H_; k0 += 32) {
      Frag16 a, bb;
      a.q[0] = *(const int4*)&s_desm[n_l][k0 + kh * 8];
      a.q[1] = *(const int4*)&s_desm[n_l][k0 + 16 + kh * 8];
      const unsigned short* bp = brow + k0 + kh * 8;
      bb.q[0] = *(const int4*)bp;
      bb.q[1] = *(const int4*)(bp + 16);
      acc = __builtin_amdgcn_wmma_f32_16x16x32_bf16(false, a.v, false, bb.v,
                                                    (short)0, acc, false, false);
    }
    const int mk = mask[b * T_ + t0 + n_l];
#pragma unroll
    for (int v = 0; v < 8; ++v)
      s_score[v + 8 * kh][t0 + n_l] = mk ? acc[v] : NEGV;
  }
  __syncthreads();

  // ---- phase 2: softmax over T (2 rows per wave) ----
  for (int rr = 0; rr < 2; ++rr) {
    int r = wave * 2 + rr;
    float mx = -3.4e38f;
    for (int c = lane; c < T_; c += 32) mx = fmaxf(mx, s_score[r][c]);
    for (int off = 16; off; off >>= 1) mx = fmaxf(mx, __shfl_xor(mx, off, 32));
    float sum = 0.0f;
    for (int c = lane; c < T_; c += 32) {
      float e = __expf(s_score[r][c] - mx);
      s_score[r][c] = e;
      sum += e;
    }
    for (int off = 16; off; off >>= 1) sum += __shfl_xor(sum, off, 32);
    float inv = 1.0f / sum;
    for (int c = lane; c < T_; c += 32) s_score[r][c] *= inv;
  }
  __syncthreads();

  // ---- phase 3: m[16][H] = alpha @ text[b]  (K = T; each wave owns 64 H cols) ----
  const v8f vzero = {0.f, 0.f, 0.f, 0.f, 0.f, 0.f, 0.f, 0.f};
  v8f macc[4] = {vzero, vzero, vzero, vzero};
  for (int k0 = 0; k0 < T_; k0 += 32) {
    __syncthreads();
    // cooperatively stage transposed 32 x H chunk of text[b].
    // lane mapping: consecutive lanes -> consecutive chunk rows r (bank-friendly
    // scatter: a wave's 32 stores span 64 consecutive bytes per j).
    for (int i = tid; i < (32 * H_) / 8; i += 256) {
      int r = i & 31;
      int c = (i >> 5) << 3;
      int4 chunk = *(const int4*)(text_bf + (size_t)(b * T_ + k0 + r) * H_ + c);
      if (k0 + 32 < T_)
        __builtin_prefetch(text_bf + (size_t)(b * T_ + k0 + 32 + r) * H_ + c, 0, 1);
      const unsigned short* ps = (const unsigned short*)&chunk;
#pragma unroll
      for (int j = 0; j < 8; ++j) sBT[(c + j) * SBT_S + r] = ps[j];
    }
    __syncthreads();
    Frag16 a;
#pragma unroll
    for (int e = 0; e < 16; ++e)
      a.u[e] = f2bf(s_score[n_l][k0 + a_koff(e, kh)]);
#pragma unroll
    for (int nt = 0; nt < 4; ++nt) {
      const int n0 = wave * 64 + nt * 16;
      Frag16 bb;
      const unsigned short* bp = &sBT[(n0 + n_l) * SBT_S + kh * 8];
      bb.q[0] = *(const int4*)bp;
      bb.q[1] = *(const int4*)(bp + 16);
      macc[nt] = __builtin_amdgcn_wmma_f32_16x16x32_bf16(false, a.v, false, bb.v,
                                                         (short)0, macc[nt], false, false);
    }
  }
  __syncthreads();
  // stash bf16(m) into s_desm for phase 4 A-fragments
#pragma unroll
  for (int nt = 0; nt < 4; ++nt) {
    const int n0 = wave * 64 + nt * 16;
#pragma unroll
    for (int v = 0; v < 8; ++v)
      s_desm[v + 8 * kh][n0 + n_l] = f2bf(macc[nt][v]);
  }
  __syncthreads();

  // ---- phase 4: s_l_p[16][1024] = m @ gate[b]^T  (K = H; M padded to 1024) ----
  float (*s_slp)[1024] = (float(*)[1024]) & s_score[0][0];
  for (int ti = 0; ti < 8; ++ti) {
    const int n0 = (wave * 8 + ti) * 16;
    const int mg = n0 + n_l;                         // M index
    const int mc = mg < M_ ? mg : (M_ - 1);
    const unsigned short* brow = gate_bf + (size_t)(b * M_ + mc) * H_;
    if (ti + 1 < 8) {
      int mg2 = n0 + 16 + n_l;
      if (mg2 >= M_) mg2 = M_ - 1;
      __builtin_prefetch(gate_bf + (size_t)(b * M_ + mg2) * H_, 0, 1);
    }
    v8f acc = vzero;
    for (int k0 = 0; k0 < H_; k0 += 32) {
      Frag16 a, bb;
      a.q[0] = *(const int4*)&s_desm[n_l][k0 + kh * 8];
      a.q[1] = *(const int4*)&s_desm[n_l][k0 + 16 + kh * 8];
      const unsigned short* bp = brow + k0 + kh * 8;
      bb.q[0] = *(const int4*)bp;
      bb.q[1] = *(const int4*)(bp + 16);
      acc = __builtin_amdgcn_wmma_f32_16x16x32_bf16(false, a.v, false, bb.v,
                                                    (short)0, acc, false, false);
    }
#pragma unroll
    for (int v = 0; v < 8; ++v)
      s_slp[v + 8 * kh][n0 + n_l] = (mg < M_) ? acc[v] : -1e30f;
  }
  __syncthreads();

  // ---- softmax over M (padded cols contribute exp->0) ----
  for (int rr = 0; rr < 2; ++rr) {
    int r = wave * 2 + rr;
    float mx = -3.4e38f;
    for (int c = lane; c < 1024; c += 32) mx = fmaxf(mx, s_slp[r][c]);
    for (int off = 16; off; off >>= 1) mx = fmaxf(mx, __shfl_xor(mx, off, 32));
    float sum = 0.0f;
    for (int c = lane; c < 1024; c += 32) {
      float e = __expf(s_slp[r][c] - mx);
      s_slp[r][c] = e;
      sum += e;
    }
    for (int off = 16; off; off >>= 1) sum += __shfl_xor(sum, off, 32);
    float inv = 1.0f / sum;
    for (int c = lane; c < 1024; c += 32) s_slp[r][c] *= inv;
  }
  __syncthreads();

  // ---- phase 5: att[16][H] = P @ wop[b]  (K = 1024 padded M) ----
  v8f aacc[4] = {vzero, vzero, vzero, vzero};
  for (int k0 = 0; k0 < 1024; k0 += 32) {
    __syncthreads();
    for (int i = tid; i < (32 * H_) / 8; i += 256) {
      int r = i & 31;
      int c = (i >> 5) << 3;
      int grow = k0 + r;
      if (grow >= M_) grow = M_ - 1;                 // P==0 there, value irrelevant
      int4 chunk = *(const int4*)(wop_bf + (size_t)(b * M_ + grow) * H_ + c);
      if (k0 + 32 < 1024) {
        int g2 = k0 + 32 + r;
        if (g2 >= M_) g2 = M_ - 1;
        __builtin_prefetch(wop_bf + (size_t)(b * M_ + g2) * H_ + c, 0, 1);
      }
      const unsigned short* ps = (const unsigned short*)&chunk;
#pragma unroll
      for (int j = 0; j < 8; ++j) sBT[(c + j) * SBT_S + r] = ps[j];
    }
    __syncthreads();
    Frag16 a;
#pragma unroll
    for (int e = 0; e < 16; ++e)
      a.u[e] = f2bf(s_slp[n_l][k0 + a_koff(e, kh)]);
#pragma unroll
    for (int nt = 0; nt < 4; ++nt) {
      const int n0 = wave * 64 + nt * 16;
      Frag16 bb;
      const unsigned short* bp = &sBT[(n0 + n_l) * SBT_S + kh * 8];
      bb.q[0] = *(const int4*)bp;
      bb.q[1] = *(const int4*)(bp + 16);
      aacc[nt] = __builtin_amdgcn_wmma_f32_16x16x32_bf16(false, a.v, false, bb.v,
                                                         (short)0, aacc[nt], false, false);
    }
  }

  // ---- phase 6: logits = sum_h (m + att) * w  (+ bvec) ----
  float rowpart[8];
#pragma unroll
  for (int v = 0; v < 8; ++v) rowpart[v] = 0.0f;
#pragma unroll
  for (int nt = 0; nt < 4; ++nt) {
    const int n0 = wave * 64 + nt * 16;
#pragma unroll
    for (int v = 0; v < 8; ++v) {
      float mf = macc[nt][v] + aacc[nt][v];
      rowpart[v] = fmaf(mf, w_f32[(size_t)(l0 + v + 8 * kh) * H_ + (n0 + n_l)],
                        rowpart[v]);
    }
  }
#pragma unroll
  for (int v = 0; v < 8; ++v) atomicAdd(&s_logit[v + 8 * kh], rowpart[v]);
  __syncthreads();
  if (tid < 16)
    out[(size_t)b * L_ + l0 + tid] = s_logit[tid] + bvec[l0 + tid];
}

// ---------------- launcher ----------------

extern "C" void kernel_launch(void* const* d_in, const int* in_sizes, int n_in,
                              void* d_out, int out_size, void* d_ws, size_t ws_size,
                              hipStream_t stream) {
  const float* text_hidden = (const float*)d_in[0];
  const int*   text_mask   = (const int*)d_in[1];
  const float* des_hidden  = (const float*)d_in[2];
  const float* wop         = (const float*)d_in[3];
  const float* p_p_h       = (const float*)d_in[4];
  const float* w_lin_W     = (const float*)d_in[5];
  const float* w_lin_b     = (const float*)d_in[6];
  const float* b_lin_W     = (const float*)d_in[7];
  const float* b_lin_b     = (const float*)d_in[8];
  const float* two_wp_W    = (const float*)d_in[9];
  float* out = (float*)d_out;

  unsigned char* ws = (unsigned char*)d_ws;
  size_t off = 0;
  auto carve = [&](size_t bytes) -> void* {
    void* p = ws + off;
    off += (bytes + 255) & ~(size_t)255;
    return p;
  };
  const int nText = B_ * T_ * H_;   // 8388608
  const int nDes  = L_ * H_;        // 4096000
  const int nBMH  = B_ * M_ * H_;   // 4096000

  unsigned short* text_bf = (unsigned short*)carve((size_t)nText * 2);
  unsigned short* des_bf  = (unsigned short*)carve((size_t)nDes * 2);
  unsigned short* gate_bf = (unsigned short*)carve((size_t)nBMH * 2);
  unsigned short* wop_bf  = (unsigned short*)carve((size_t)nBMH * 2);
  float* w_f32 = (float*)carve((size_t)nDes * 4);
  float* bvec  = (float*)carve((size_t)L_ * 4);

  k_f32_to_bf16<<<(nText + 255) / 256, 256, 0, stream>>>(text_hidden, text_bf, nText);
  k_f32_to_bf16<<<(nDes + 255) / 256, 256, 0, stream>>>(des_hidden, des_bf, nDes);
  k_gate_wop<<<(nBMH + 255) / 256, 256, 0, stream>>>(p_p_h, two_wp_W, wop,
                                                     gate_bf, wop_bf, nBMH);
  k_wlin<<<L_, 256, 0, stream>>>(des_hidden, w_lin_W, w_lin_b, w_f32);
  k_bvec<<<(L_ + 255) / 256, 256, 0, stream>>>(des_hidden, b_lin_W, b_lin_b, bvec);

  fused_att_kernel<<<B_ * (L_ / 16), 256, 0, stream>>>(
      text_bf, text_mask, des_bf, gate_bf, wop_bf, w_f32, bvec, out);
}
